// Branch_3_36386962932307
// MI455X (gfx1250) — matrix operand order repaired and checked by reading, not compiled, not run on hardware
//
#include <hip/hip_runtime.h>
#include <hip/hip_bf16.h>

typedef __bf16 bf16_t;
typedef __attribute__((ext_vector_type(16))) bf16_t v16bf;
typedef __attribute__((ext_vector_type(8)))  bf16_t v8bf;
typedef __attribute__((ext_vector_type(8)))  float  v8f;

#define DEV __device__ __forceinline__

DEV bf16_t f2bf(float x) {
  union { float f; unsigned u; } v; v.f = x;
  unsigned r = v.u + 0x7FFFu + ((v.u >> 16) & 1u);
  unsigned short h = (unsigned short)(r >> 16);
  union { unsigned short s; bf16_t b; } o; o.s = h; return o.b;
}
DEV float sigmoidf_(float x) { return 1.0f / (1.0f + __expf(-x)); }
DEV float siluf_(float x)    { return x * sigmoidf_(x); }
DEV float softplusf_(float x){ return (x > 20.0f) ? x : log1pf(__expf(x)); }

// ---------------------------------------------------------------------------
// WMMA fragment loads (ISA §7.12.2 wave32 layouts).
// A row-major MxK: lane sub=lane&15 owns row m0+sub; half=lane>>4 picks the
//   K-interleave: elements 0..7 = K kb+8*half.., elements 8..15 = kb+16+8*half..
// B = W^T with W row-major NxK: lane owns W row n0+sub, same K-interleave as A:
//   elements 0..7 = kb+8*half.., elements 8..15 = kb+16+8*half..
// ---------------------------------------------------------------------------
DEV v16bf load_afrag(const bf16_t* __restrict__ arow, int kb, int half) {
  v8bf a0 = *(const v8bf*)(arow + kb + 8 * half);
  v8bf a1 = *(const v8bf*)(arow + kb + 16 + 8 * half);
  v16bf av;
#pragma unroll
  for (int i = 0; i < 8; ++i) { av[i] = a0[i]; av[i + 8] = a1[i]; }
  return av;
}
DEV v16bf load_bfrag(const bf16_t* __restrict__ wrow, int kb, int half) {
  v8bf b0 = *(const v8bf*)(wrow + kb + 8 * half);
  v8bf b1 = *(const v8bf*)(wrow + kb + 16 + 8 * half);
  v16bf bv;
#pragma unroll
  for (int i = 0; i < 8; ++i) { bv[i] = b0[i]; bv[i + 8] = b1[i]; }
  return bv;
}

// Single 16x16 tile (used only for the padded x_proj GEMM, N=80)
DEV v8f wmma_tile_1x1(const bf16_t* __restrict__ A, int lda,
                      const bf16_t* __restrict__ W, int ldw,
                      int m0, int n0, int K) {
  const int lane = threadIdx.x & 31, sub = lane & 15, half = lane >> 4;
  const bf16_t* arow = A + (size_t)(m0 + sub) * (size_t)lda;
  const bf16_t* wrow = W + (size_t)(n0 + sub) * (size_t)ldw;
  v8f acc = {};
  for (int kb = 0; kb < K; kb += 32) {
    v16bf av = load_afrag(arow, kb, half);
    v16bf bv = load_bfrag(wrow, kb, half);
    acc = __builtin_amdgcn_wmma_f32_16x16x32_bf16(
        false, av, false, bv, (short)0, acc, false, false);
  }
  return acc;
}

// 2x2 register-blocked 32x32 macro-tile: 4 WMMAs per K-step from 2 A-frags +
// 2 B-frags -> half the vmem bytes / load instructions per WMMA vs 1x1.
struct Acc4 { v8f a00, a01, a10, a11; };

DEV Acc4 wmma_tile_2x2(const bf16_t* __restrict__ A, int lda,
                       const bf16_t* __restrict__ W, int ldw,
                       int m0, int n0, int K) {
  const int lane = threadIdx.x & 31, sub = lane & 15, half = lane >> 4;
  const bf16_t* arow0 = A + (size_t)(m0 + sub) * (size_t)lda;
  const bf16_t* arow1 = arow0 + (size_t)16 * (size_t)lda;
  const bf16_t* wrow0 = W + (size_t)(n0 + sub) * (size_t)ldw;
  const bf16_t* wrow1 = wrow0 + (size_t)16 * (size_t)ldw;
  Acc4 t; t.a00 = (v8f){}; t.a01 = (v8f){}; t.a10 = (v8f){}; t.a11 = (v8f){};
  for (int kb = 0; kb < K; kb += 32) {
    v16bf av0 = load_afrag(arow0, kb, half);
    v16bf av1 = load_afrag(arow1, kb, half);
    v16bf bv0 = load_bfrag(wrow0, kb, half);
    v16bf bv1 = load_bfrag(wrow1, kb, half);
    t.a00 = __builtin_amdgcn_wmma_f32_16x16x32_bf16(false, av0, false, bv0, (short)0, t.a00, false, false);
    t.a01 = __builtin_amdgcn_wmma_f32_16x16x32_bf16(false, av0, false, bv1, (short)0, t.a01, false, false);
    t.a10 = __builtin_amdgcn_wmma_f32_16x16x32_bf16(false, av1, false, bv0, (short)0, t.a10, false, false);
    t.a11 = __builtin_amdgcn_wmma_f32_16x16x32_bf16(false, av1, false, bv1, (short)0, t.a11, false, false);
  }
  return t;
}

// Epilogue + store of one 16x16 tile. C/D layout: acc[r] -> row m0+r+8*half,
// col n0+(lane&15).
template <int EPI, bool HAS_BIAS, bool OUT_BF>
DEV void store_tile(v8f acc, const float* __restrict__ bias,
                    void* __restrict__ out, int ldo, int m0, int n0) {
  const int lane = threadIdx.x & 31, sub = lane & 15, half = lane >> 4;
  const int col = n0 + sub;
  const float bv = HAS_BIAS ? bias[col] : 0.0f;
#pragma unroll
  for (int r = 0; r < 8; ++r) {
    int row = m0 + r + 8 * half;
    float v = acc[r] + bv;
    if (EPI == 1) v = siluf_(v);
    else if (EPI == 2) v = softplusf_(v);
    if (OUT_BF) ((bf16_t*)out)[(size_t)row * ldo + col] = f2bf(v);
    else        ((float *)out)[(size_t)row * ldo + col] = v;
  }
}

// ---------------------------------------------------------------------------
// Generic 2x2-blocked GEMM: EPI 0=none 1=silu 2=softplus. M,N multiples of 32.
// ---------------------------------------------------------------------------
template <int EPI, bool HAS_BIAS, bool OUT_BF>
__global__ void gemm_bf16_kernel(const bf16_t* __restrict__ A, int lda,
                                 const bf16_t* __restrict__ W, int ldw,
                                 const float* __restrict__ bias,
                                 void* __restrict__ out, int ldo,
                                 int M, int N, int K) {
  int wave = blockIdx.x * (blockDim.x >> 5) + (threadIdx.x >> 5);
  int tn = N >> 5;
  int total = (M >> 5) * tn;
  if (wave >= total) return;
  int m0 = (wave / tn) << 5, n0 = (wave % tn) << 5;
  Acc4 t = wmma_tile_2x2(A, lda, W, ldw, m0, n0, K);
  store_tile<EPI, HAS_BIAS, OUT_BF>(t.a00, bias, out, ldo, m0,      n0);
  store_tile<EPI, HAS_BIAS, OUT_BF>(t.a01, bias, out, ldo, m0,      n0 + 16);
  store_tile<EPI, HAS_BIAS, OUT_BF>(t.a10, bias, out, ldo, m0 + 16, n0);
  store_tile<EPI, HAS_BIAS, OUT_BF>(t.a11, bias, out, ldo, m0 + 16, n0 + 16);
}

// in_proj GEMM: M=8192, N=2048, K=1024; cols <1024 -> xs, >=1024 -> silu(z).
// n0 is a multiple of 32 so each 16-wide subtile is entirely on one side.
__global__ void gemm_xz_kernel(const bf16_t* __restrict__ A,
                               const bf16_t* __restrict__ W,
                               float* __restrict__ xs, float* __restrict__ zs) {
  const int tn = 2048 >> 5;
  int wave = blockIdx.x * (blockDim.x >> 5) + (threadIdx.x >> 5);
  int total = (8192 >> 5) * tn;
  if (wave >= total) return;
  int m0 = (wave / tn) << 5, n0 = (wave % tn) << 5;
  Acc4 t = wmma_tile_2x2(A, 1024, W, 1024, m0, n0, 1024);
  const int lane = threadIdx.x & 31, sub = lane & 15, half = lane >> 4;
  v8f accs[4] = { t.a00, t.a01, t.a10, t.a11 };
#pragma unroll
  for (int q = 0; q < 4; ++q) {
    int col = n0 + sub + (q & 1) * 16;
    int mb  = m0 + (q >> 1) * 16;
#pragma unroll
    for (int r = 0; r < 8; ++r) {
      int row = mb + r + 8 * half;
      float v = accs[q][r];
      if (col < 1024) xs[(size_t)row * 1024 + col] = v;
      else            zs[(size_t)row * 1024 + (col - 1024)] = siluf_(v);
    }
  }
}

// x_proj GEMM: N padded to 80 (1x1 tiles); cols 0..63 -> dt(bf16), 64..67 -> B/C
__global__ void gemm_xproj_kernel(const bf16_t* __restrict__ A,
                                  const bf16_t* __restrict__ W,
                                  bf16_t* __restrict__ dtb,
                                  float* __restrict__ BC) {
  const int tn = 80 >> 4;
  int wave = blockIdx.x * (blockDim.x >> 5) + (threadIdx.x >> 5);
  int total = (8192 >> 4) * tn;
  if (wave >= total) return;
  int m0 = (wave / tn) << 4, n0 = (wave % tn) << 4;
  v8f acc = wmma_tile_1x1(A, 1024, W, 1024, m0, n0, 1024);
  const int lane = threadIdx.x & 31, sub = lane & 15, half = lane >> 4;
  const int col = n0 + sub;
#pragma unroll
  for (int r = 0; r < 8; ++r) {
    int row = m0 + r + 8 * half;
    float v = acc[r];
    if (col < 64)       dtb[(size_t)row * 64 + col] = f2bf(v);
    else if (col < 68)  BC [(size_t)row * 4 + (col - 64)] = v;
  }
}

// ---------------------------------------------------------------------------
// Elementwise / scan / LN kernels
// ---------------------------------------------------------------------------
__global__ void cvt_bf16_kernel(const float* __restrict__ s,
                                bf16_t* __restrict__ d, int n) {
  int i = blockIdx.x * blockDim.x + threadIdx.x;
  if (i < n) d[i] = f2bf(s[i]);
}

__global__ void cvt_pad_xproj_kernel(const float* __restrict__ s,
                                     bf16_t* __restrict__ d) {
  int i = blockIdx.x * blockDim.x + threadIdx.x;
  if (i >= 80 * 1024) return;
  d[i] = (i < 68 * 1024) ? f2bf(s[i]) : f2bf(0.0f);
}

// causal depthwise conv (kernel 4) along L, + bias + SiLU; f32 & bf16 out
__global__ void conv_silu_kernel(const float* __restrict__ xs,
                                 const float* __restrict__ cw,
                                 const float* __restrict__ cb,
                                 float* __restrict__ xa,
                                 bf16_t* __restrict__ xab) {
  int idx = blockIdx.x * blockDim.x + threadIdx.x;
  if (idx >= 16 * 512 * 1024) return;
  int d = idx & 1023, l = (idx >> 10) & 511, b = idx >> 19;
  float acc = cb[d];
  size_t base = (size_t)(b * 512) * 1024 + d;
#pragma unroll
  for (int j = 0; j < 4; ++j) {
    int ls = l - 3 + j;
    if (ls >= 0) acc += cw[d * 4 + j] * xs[base + (size_t)ls * 1024];
  }
  float v = siluf_(acc);
  xa[idx] = v;
  xab[idx] = f2bf(v);
}

// selective-scan: one thread per (b,d); h[n=0..1] in regs; fused D-skip + gate
__global__ void scan_kernel(const float* __restrict__ delta,
                            const float* __restrict__ xa,
                            const float* __restrict__ BC,
                            const float* __restrict__ zs,
                            const float* __restrict__ A_log,
                            const float* __restrict__ Dp,
                            bf16_t* __restrict__ yb) {
  int b = blockIdx.x >> 2;
  int d = ((blockIdx.x & 3) << 8) | threadIdx.x;
  float A0 = -__expf(A_log[d * 2 + 0]);
  float A1 = -__expf(A_log[d * 2 + 1]);
  float Dd = Dp[d];
  float h0 = 0.f, h1 = 0.f;
  size_t rowbase = (size_t)b * 512;
  for (int l = 0; l < 512; ++l) {
    size_t row = rowbase + l;
    size_t off = row * 1024 + d;
    float dl = delta[off];
    float xv = xa[off];
    const float* bc = BC + row * 4;
    float du = dl * xv;
    h0 = __expf(dl * A0) * h0 + du * bc[0];
    h1 = __expf(dl * A1) * h1 + du * bc[1];
    float y = h0 * bc[2] + h1 * bc[3] + xv * Dd;
    y *= zs[off];                     // zs holds silu(z)
    yb[off] = f2bf(y);
  }
}

// LayerNorm over 1024 + transposed bf16 store: (b,c,s) -> (b,s,c)
__global__ void ln_transpose_kernel(const float* __restrict__ y,
                                    const float* __restrict__ g,
                                    const float* __restrict__ bb,
                                    bf16_t* __restrict__ ynt) {
  int row = blockIdx.x;                  // b*512 + c
  int b = row >> 9, c = row & 511;
  const float* yr = y + (size_t)row * 1024;
  float s = 0.f, s2 = 0.f;
  for (int i = threadIdx.x; i < 1024; i += 256) {
    float v = yr[i]; s += v; s2 += v * v;
  }
  __shared__ float sh[256], sh2[256];
  sh[threadIdx.x] = s; sh2[threadIdx.x] = s2;
  __syncthreads();
  for (int off = 128; off > 0; off >>= 1) {
    if (threadIdx.x < off) {
      sh[threadIdx.x]  += sh[threadIdx.x + off];
      sh2[threadIdx.x] += sh2[threadIdx.x + off];
    }
    __syncthreads();
  }
  float mu  = sh[0]  * (1.0f / 1024.0f);
  float var = sh2[0] * (1.0f / 1024.0f) - mu * mu;
  float rs = rsqrtf(var + 1e-5f);
  for (int i = threadIdx.x; i < 1024; i += 256) {
    float v = (yr[i] - mu) * rs * g[i] + bb[i];
    ynt[((size_t)b * 1024 + i) * 512 + c] = f2bf(v);
  }
}

// (b,c,t) -> (b,t,c) bf16 transpose
__global__ void transpose_bct_kernel(const bf16_t* __restrict__ src,
                                     bf16_t* __restrict__ dst) {
  int idx = blockIdx.x * blockDim.x + threadIdx.x;
  if (idx >= 16 * 512 * 1024) return;
  int t = idx & 1023, c = (idx >> 10) & 511, b = idx >> 19;
  dst[((size_t)(b * 1024) + t) * 512 + c] = src[idx];
}

// d_out[b,o,s] = ybranch[b,s,o] + r2[b,s,o]
__global__ void final_add_kernel(const float* __restrict__ yb,
                                 const float* __restrict__ r2,
                                 float* __restrict__ out) {
  int idx = blockIdx.x * blockDim.x + threadIdx.x;
  if (idx >= 16 * 512 * 1024) return;
  int s = idx & 1023, o = (idx >> 10) & 511, b = idx >> 19;
  size_t src = ((size_t)(b * 1024) + s) * 512 + o;
  out[idx] = yb[src] + r2[src];
}

// ---------------------------------------------------------------------------
extern "C" void kernel_launch(void* const* d_in, const int* in_sizes, int n_in,
                              void* d_out, int out_size, void* d_ws, size_t ws_size,
                              hipStream_t stream) {
  (void)in_sizes; (void)n_in; (void)out_size; (void)ws_size;
  const float* x        = (const float*)d_in[0];
  const float* in_proj  = (const float*)d_in[1];
  const float* conv_w   = (const float*)d_in[2];
  const float* conv_b   = (const float*)d_in[3];
  const float* x_proj   = (const float*)d_in[4];
  const float* dt_projw = (const float*)d_in[5];
  const float* dt_projb = (const float*)d_in[6];
  const float* A_log    = (const float*)d_in[7];
  const float* D_ssm    = (const float*)d_in[8];
  const float* out_proj = (const float*)d_in[9];
  const float* ln_g     = (const float*)d_in[10];
  const float* ln_b     = (const float*)d_in[11];
  const float* lin3_w   = (const float*)d_in[12];
  const float* lin3_b   = (const float*)d_in[13];
  const float* linsp_w  = (const float*)d_in[14];
  const float* linsp_b  = (const float*)d_in[15];
  const float* linres_w = (const float*)d_in[16];
  const float* linres_b = (const float*)d_in[17];
  float* outp = (float*)d_out;

  char* w = (char*)d_ws;
  size_t off = 0;
  auto take = [&](size_t bytes) -> char* {
    char* p = w + off;
    off += (bytes + 255) & ~(size_t)255;
    return p;
  };
  const size_t NE = 16u * 512u * 1024u;           // 8,388,608
  bf16_t* x_bf   = (bf16_t*)take(NE * 2);
  bf16_t* w_inpr = (bf16_t*)take((size_t)2048 * 1024 * 2);
  bf16_t* w_xpad = (bf16_t*)take((size_t)80 * 1024 * 2);
  bf16_t* w_dtpr = (bf16_t*)take((size_t)1024 * 64 * 2);
  bf16_t* w_outp = (bf16_t*)take((size_t)1024 * 1024 * 2);
  bf16_t* w_lin3 = (bf16_t*)take((size_t)512 * 512 * 2);
  bf16_t* w_lsp  = (bf16_t*)take((size_t)1024 * 1024 * 2);
  bf16_t* w_lres = (bf16_t*)take((size_t)512 * 512 * 2);
  float* xs_f    = (float*)take(NE * 4);          // pre-conv xs; later r2
  float* zs_f    = (float*)take(NE * 4);          // silu(z)
  float* xa_f    = (float*)take(NE * 4);          // conv+silu activation
  float* delta_f = (float*)take(NE * 4);          // delta; later y-branch
  float* out_f   = (float*)take(NE * 4);          // out_proj result (pre-LN)
  bf16_t* xab_bf = (bf16_t*)take(NE * 2);         // conv act bf16; later r1
  bf16_t* y_bf   = (bf16_t*)take(NE * 2);         // gated SSM output bf16
  bf16_t* ynt_bf = (bf16_t*)take(NE * 2);         // LN output, (b,s,c)
  bf16_t* r1t_bf = (bf16_t*)take(NE * 2);         // r1 transposed, (b,t,c)
  bf16_t* dt_bf  = (bf16_t*)take((size_t)8192 * 64 * 2);
  float*  bc_f   = (float*)take((size_t)8192 * 4 * 4);

  // ---- weight / input conversion to bf16 ----
  cvt_bf16_kernel<<<32768, 256, 0, stream>>>(x, x_bf, (int)NE);
  cvt_bf16_kernel<<<8192, 256, 0, stream>>>(in_proj, w_inpr, 2048 * 1024);
  cvt_pad_xproj_kernel<<<320, 256, 0, stream>>>(x_proj, w_xpad);
  cvt_bf16_kernel<<<256, 256, 0, stream>>>(dt_projw, w_dtpr, 1024 * 64);
  cvt_bf16_kernel<<<4096, 256, 0, stream>>>(out_proj, w_outp, 1024 * 1024);
  cvt_bf16_kernel<<<1024, 256, 0, stream>>>(lin3_w, w_lin3, 512 * 512);
  cvt_bf16_kernel<<<4096, 256, 0, stream>>>(linsp_w, w_lsp, 1024 * 1024);
  cvt_bf16_kernel<<<1024, 256, 0, stream>>>(linres_w, w_lres, 512 * 512);

  // ---- Mamba branch ----
  gemm_xz_kernel<<<2048, 256, 0, stream>>>(x_bf, w_inpr, xs_f, zs_f);
  conv_silu_kernel<<<32768, 256, 0, stream>>>(xs_f, conv_w, conv_b, xa_f, xab_bf);
  gemm_xproj_kernel<<<320, 256, 0, stream>>>(xab_bf, w_xpad, dt_bf, bc_f);
  // residual spatial GEMM (silu, bf16 out) reusing xab_bf region as r1
  gemm_bf16_kernel<1, true, true><<<1024, 256, 0, stream>>>(
      x_bf, 1024, w_lsp, 1024, linsp_b, xab_bf, 1024, 8192, 1024, 1024);
  transpose_bct_kernel<<<32768, 256, 0, stream>>>(xab_bf, r1t_bf);
  // delta = softplus(dt @ dt_proj^T + b)
  gemm_bf16_kernel<2, true, false><<<1024, 256, 0, stream>>>(
      dt_bf, 64, w_dtpr, 64, dt_projb, delta_f, 1024, 8192, 1024, 64);
  scan_kernel<<<64, 256, 0, stream>>>(delta_f, xa_f, bc_f, zs_f, A_log, D_ssm, y_bf);
  gemm_bf16_kernel<0, false, false><<<1024, 256, 0, stream>>>(
      y_bf, 1024, w_outp, 1024, nullptr, out_f, 1024, 8192, 1024, 1024);
  ln_transpose_kernel<<<8192, 256, 0, stream>>>(out_f, ln_g, ln_b, ynt_bf);
  // lin3 (silu) -> y-branch (b,s,o), reusing delta_f
  gemm_bf16_kernel<1, true, false><<<1024, 256, 0, stream>>>(
      ynt_bf, 512, w_lin3, 512, lin3_b, delta_f, 512, 16384, 512, 512);
  // linres (silu) -> r2 (b,s,o), reusing xs_f
  gemm_bf16_kernel<1, true, false><<<1024, 256, 0, stream>>>(
      r1t_bf, 512, w_lres, 512, linres_b, xs_f, 512, 16384, 512, 512);
  final_add_kernel<<<32768, 256, 0, stream>>>(delta_f, xs_f, outp);
}
